// GNNEncoder_47150150975506
// MI455X (gfx1250) — matrix operand order repaired and checked by reading, not compiled, not run on hardware
//
#include <hip/hip_runtime.h>
#include <hip/hip_bf16.h>
#include <math.h>

// ---------------------------------------------------------------------------
// CDNA5 (gfx1250) WMMA types
// ---------------------------------------------------------------------------
typedef __attribute__((ext_vector_type(16))) _Float16 v16h;
typedef __attribute__((ext_vector_type(8)))  float    v8f;

#define WMMA16x16x32(a, b, c) \
  __builtin_amdgcn_wmma_f32_16x16x32_f16(false, (a), false, (b), (short)0, (c), false, false)

// Model constants
#define BB   8
#define TT   65568
#define KWIN 64
#define STR  32
#define LL   2048          // encoded frames
#define DD   512
#define HH   32
#define NNODE (BB * LL)    // 16384
#define KNN8 8

// ---------------------------------------------------------------------------
// Small utility kernels
// ---------------------------------------------------------------------------
__global__ void k_f32_to_f16(const float* __restrict__ in, _Float16* __restrict__ out, long n) {
  long i = (long)blockIdx.x * blockDim.x + threadIdx.x;
  if (i < n) out[i] = (_Float16)in[i];
}

__global__ void k_fill_f32(float* p, float v, long n) {
  long i = (long)blockIdx.x * blockDim.x + threadIdx.x;
  if (i < n) p[i] = v;
}

__global__ void k_rsqrt_inplace(float* p, long n) {
  long i = (long)blockIdx.x * blockDim.x + threadIdx.x;
  if (i < n) p[i] = rsqrtf(p[i]);
}

// Reorder conv weight [cout][cin][3] -> A [cout][3*cin] with A[co][k*cin+ci]
__global__ void k_conv_w_reorder(const float* __restrict__ w, _Float16* __restrict__ A,
                                 int cout, int cin) {
  long n = (long)cout * 3 * cin;
  long i = (long)blockIdx.x * blockDim.x + threadIdx.x;
  if (i >= n) return;
  int threecin = 3 * cin;
  int co = (int)(i / threecin);
  int r  = (int)(i - (long)co * threecin);
  int k  = r / cin;
  int ci = r - k * cin;
  A[i] = (_Float16)w[((long)co * cin + ci) * 3 + k];
}

// ConvTranspose weight [cin][cout][2] -> A_k [cout][cin] (k = 0,1)
__global__ void k_ct_w_reorder(const float* __restrict__ w, _Float16* __restrict__ A0,
                               _Float16* __restrict__ A1, int cin, int cout) {
  long n = (long)cout * cin;
  long i = (long)blockIdx.x * blockDim.x + threadIdx.x;
  if (i >= n) return;
  int co = (int)(i / cin);
  int ci = (int)(i - (long)co * cin);
  const float* p = w + ((long)ci * cout + co) * 2;
  A0[i] = (_Float16)p[0];
  A1[i] = (_Float16)p[1];
}

// Fold BatchNorm (+conv bias) into scale/shift: y = s*conv + t
__global__ void k_bn_fold(const float* g, const float* bb, const float* m, const float* v,
                          const float* bias, float* scale, float* shift, int c) {
  int i = blockIdx.x * blockDim.x + threadIdx.x;
  if (i >= c) return;
  float s = g[i] * rsqrtf(v[i] + 1e-5f);
  scale[i] = s;
  shift[i] = bb[i] + s * (bias[i] - m[i]);
}

// ---------------------------------------------------------------------------
// WMMA fragment loaders (contiguous k order, wave32)
//   A 16x32 f16: lane m (0-15) = row m, K 0-15; lane m+16: K 16-31
//   B 32x16 f16: lane n (0-15) = col n, K 0-15; lane n+16: col n, K 16-31
// ---------------------------------------------------------------------------
__device__ __forceinline__ v16h load_a_frag(const _Float16* __restrict__ ap) {
  // ap already offset to row base + k0 + abase
  v16h a;
#pragma unroll
  for (int i = 0; i < 8; ++i) a[i] = ap[i];
#pragma unroll
  for (int i = 0; i < 8; ++i) a[8 + i] = ap[8 + i];
  return a;
}

// ---------------------------------------------------------------------------
// WMMA GEMM: C = A * B^T   (A row-major [M,K], B row-major [N,K])
// One 16 x (16*NBLK) C strip per wave: A fragment reused across NBLK WMMAs.
// Requires K % 32 == 0, M % 16 == 0, N % (16*NBLK) == 0.
// ---------------------------------------------------------------------------
template <int NBLK>
__global__ __launch_bounds__(256)
void k_gemm_nt(const _Float16* __restrict__ A, int lda, long sA,
               const _Float16* __restrict__ Bm, int ldb, long sB,
               float* __restrict__ Cf32, _Float16* __restrict__ Cf16, int ldc, long sC,
               int M, int N, int K) {
  int wave = threadIdx.x >> 5;
  int lane = threadIdx.x & 31;
  int tn = blockIdx.x * (blockDim.x >> 5) + wave;
  int tm = blockIdx.y;
  int bz = blockIdx.z;
  int col0 = tn * 16 * NBLK;
  if (col0 >= N) return;
  const _Float16* Ab = A + (long)bz * sA;
  const _Float16* Bb = Bm + (long)bz * sB;
  int row0 = tm * 16;
  int m = lane & 15, n = lane & 15;
  int abase = (lane < 16) ? 0 : 16;
  int bbase = (lane < 16) ? 0 : 16;
  v8f acc[NBLK];
#pragma unroll
  for (int j = 0; j < NBLK; ++j) acc[j] = (v8f){};
  for (int k0 = 0; k0 < K; k0 += 32) {
    v16h a = load_a_frag(Ab + (long)(row0 + m) * lda + k0 + abase);
#pragma unroll
    for (int j = 0; j < NBLK; ++j) {
      v16h bt;
      const _Float16* bp = Bb + (long)(col0 + j * 16 + n) * ldb + k0 + bbase;
#pragma unroll
      for (int i = 0; i < 16; ++i) bt[i] = bp[i];
      acc[j] = WMMA16x16x32(a, bt, acc[j]);
    }
  }
  int mb = (lane < 16) ? 0 : 8;
#pragma unroll
  for (int j = 0; j < NBLK; ++j) {
#pragma unroll
    for (int r = 0; r < 8; ++r) {
      int row = row0 + mb + r;
      float v = acc[j][r];
      long idx = (long)bz * sC + (long)row * ldc + col0 + j * 16 + n;
      if (Cf32) Cf32[idx] = v;
      if (Cf16) Cf16[idx] = (_Float16)v;
    }
  }
}

// ---------------------------------------------------------------------------
// WMMA GEMM: C = A * B   (A row-major [M,K], B row-major [K,N])
// Epilogue: optional per-row scale/shift, relu, strided columns (for convT).
// ---------------------------------------------------------------------------
template <int NBLK>
__global__ __launch_bounds__(256)
void k_gemm_nn(const _Float16* __restrict__ A, int lda, long sA,
               const _Float16* __restrict__ Bm, int ldb, long sB,
               float* __restrict__ Cf32, _Float16* __restrict__ Cf16, int ldc, long sC,
               const float* __restrict__ scale, const float* __restrict__ shift, int relu,
               int colMul, int colAdd, int M, int N, int K) {
  int wave = threadIdx.x >> 5;
  int lane = threadIdx.x & 31;
  int tn = blockIdx.x * (blockDim.x >> 5) + wave;
  int tm = blockIdx.y;
  int bz = blockIdx.z;
  int col0 = tn * 16 * NBLK;
  if (col0 >= N) return;
  const _Float16* Ab = A + (long)bz * sA;
  const _Float16* Bb = Bm + (long)bz * sB;
  int row0 = tm * 16;
  int m = lane & 15, n = lane & 15;
  int abase = (lane < 16) ? 0 : 16;
  int bbase = (lane < 16) ? 0 : 16;
  v8f acc[NBLK];
#pragma unroll
  for (int j = 0; j < NBLK; ++j) acc[j] = (v8f){};
  for (int k0 = 0; k0 < K; k0 += 32) {
    v16h a = load_a_frag(Ab + (long)(row0 + m) * lda + k0 + abase);
    const _Float16* bp0 = Bb + (long)(k0 + bbase) * ldb + col0 + n;
#pragma unroll
    for (int j = 0; j < NBLK; ++j) {
      v16h bt;
      const _Float16* bp = bp0 + j * 16;
#pragma unroll
      for (int i = 0; i < 16; ++i) bt[i] = bp[(long)i * ldb];
      acc[j] = WMMA16x16x32(a, bt, acc[j]);
    }
  }
  int mb = (lane < 16) ? 0 : 8;
#pragma unroll
  for (int j = 0; j < NBLK; ++j) {
#pragma unroll
    for (int r = 0; r < 8; ++r) {
      int row = row0 + mb + r;
      float v = acc[j][r];
      if (scale) v = v * scale[row] + (shift ? shift[row] : 0.f);
      else if (shift) v += shift[row];
      if (relu) v = v > 0.f ? v : 0.f;
      int col = (col0 + j * 16 + n) * colMul + colAdd;
      long idx = (long)bz * sC + (long)row * ldc + col;
      if (Cf32) Cf32[idx] = v;
      if (Cf16) Cf16[idx] = (_Float16)v;
    }
  }
}

// ---------------------------------------------------------------------------
// Implicit-GEMM conv1d (kernel=3, pad=1) + folded BN + ReLU, f16 in/out.
// A = reordered weights [cout][3*cin] (shared across all column tiles ->
// NBLK=4 blocking amortizes the entire weight stream).
// ---------------------------------------------------------------------------
template <int NBLK>
__global__ __launch_bounds__(256)
void k_conv3(const _Float16* __restrict__ X, long sX,
             const _Float16* __restrict__ W,
             const float* __restrict__ scale, const float* __restrict__ shift,
             _Float16* __restrict__ Y, long sY,
             int cin, int cout, int Llen) {
  int wave = threadIdx.x >> 5;
  int lane = threadIdx.x & 31;
  int tn = blockIdx.x * (blockDim.x >> 5) + wave;
  int tm = blockIdx.y;
  int bz = blockIdx.z;
  int col0 = tn * 16 * NBLK;
  if (col0 >= Llen) return;
  const _Float16* Xb = X + (long)bz * sX;
  int row0 = tm * 16;
  int m = lane & 15, n = lane & 15;
  int abase = (lane < 16) ? 0 : 16;
  int bbase = (lane < 16) ? 0 : 16;
  int K = 3 * cin;
  const _Float16* wp = W + (long)(row0 + m) * K;
  v8f acc[NBLK];
#pragma unroll
  for (int j = 0; j < NBLK; ++j) acc[j] = (v8f){};
  for (int k0 = 0; k0 < K; k0 += 32) {
    v16h a = load_a_frag(wp + k0 + abase);
    int kk = k0 + bbase;             // run of 16 K-indices, never crosses cin boundary
    int kidx = kk / cin;             // tap 0..2
    int ci0 = kk - kidx * cin;
    const _Float16* xcol = Xb + (long)ci0 * Llen;
#pragma unroll
    for (int j = 0; j < NBLK; ++j) {
      v16h bt = {};
      int l = col0 + j * 16 + n + kidx - 1;   // pad = 1
      if (l >= 0 && l < Llen) {
        const _Float16* xp = xcol + l;
#pragma unroll
        for (int i = 0; i < 16; ++i) bt[i] = xp[(long)i * Llen];
      }
      acc[j] = WMMA16x16x32(a, bt, acc[j]);
    }
  }
  int mb = (lane < 16) ? 0 : 8;
#pragma unroll
  for (int j = 0; j < NBLK; ++j) {
#pragma unroll
    for (int r = 0; r < 8; ++r) {
      int co = row0 + mb + r;
      float v = acc[j][r] * scale[co] + shift[co];
      v = v > 0.f ? v : 0.f;
      Y[(long)bz * sY + (long)co * Llen + col0 + j * 16 + n] = (_Float16)v;
    }
  }
}

// ---------------------------------------------------------------------------
// kNN helpers
// ---------------------------------------------------------------------------
__global__ void k_sq_rows(const _Float16* __restrict__ F, float* __restrict__ sq, int K) {
  int i = blockIdx.x * blockDim.x + threadIdx.x;
  if (i >= NNODE) return;
  const _Float16* p = F + (long)i * K;
  float acc = 0.f;
  for (int d = 0; d < K; ++d) { float x = (float)p[d]; acc += x * x; }
  sq[i] = acc;
}

__global__ void k_topk(const float* __restrict__ G, const float* __restrict__ sq,
                       int* __restrict__ nbr, int b) {
  int i = blockIdx.x * blockDim.x + threadIdx.x;
  if (i >= LL) return;
  float bd[KNN8]; int bi[KNN8];
#pragma unroll
  for (int k = 0; k < KNN8; ++k) { bd[k] = 3.0e38f; bi[k] = 0; }
  float sqi = sq[b * LL + i];
  const float* g = G + (long)i * LL;
  for (int j = 0; j < LL; ++j) {
    if (j == i) continue;
    float d = sqi + sq[b * LL + j] - 2.f * g[j];
    if (d < bd[KNN8 - 1]) {
      int p = KNN8 - 1;
      while (p > 0 && bd[p - 1] > d) { bd[p] = bd[p - 1]; bi[p] = bi[p - 1]; --p; }
      bd[p] = d; bi[p] = b * LL + j;     // global node index
    }
  }
  int* o = nbr + ((long)b * LL + i) * KNN8;
#pragma unroll
  for (int k = 0; k < KNN8; ++k) o[k] = bi[k];
}

__global__ void k_deg_accum(const int* __restrict__ nbr, float* __restrict__ deg, long n) {
  long i = (long)blockIdx.x * blockDim.x + threadIdx.x;
  if (i < n) atomicAdd(&deg[nbr[i]], 1.f);
}

// ---------------------------------------------------------------------------
// GCN aggregation
// ---------------------------------------------------------------------------
__global__ void k_gcn_init(const float* __restrict__ h, const float* __restrict__ dinv,
                           const float* __restrict__ bias, float* __restrict__ agg, int Fdim) {
  long idx = (long)blockIdx.x * blockDim.x + threadIdx.x;
  long total = (long)NNODE * Fdim;
  if (idx >= total) return;
  int i = (int)(idx / Fdim);
  int f = (int)(idx - (long)i * Fdim);
  float di = dinv[i];
  agg[idx] = h[idx] * di * di + bias[f];
}

__global__ void k_gcn_edges(const float* __restrict__ h, const float* __restrict__ dinv,
                            const int* __restrict__ nbr, float* __restrict__ agg, int Fdim) {
  long idx = (long)blockIdx.x * blockDim.x + threadIdx.x;
  long total = (long)NNODE * Fdim;
  if (idx >= total) return;
  int i = (int)(idx / Fdim);
  int f = (int)(idx - (long)i * Fdim);
  float di = dinv[i];
  float hi = h[idx];
  float acc = 0.f;
#pragma unroll
  for (int k = 0; k < KNN8; ++k) {
    int j = nbr[(long)i * KNN8 + k];
    float cc = di * dinv[j];
    acc += h[(long)j * Fdim + f] * cc;
    atomicAdd(&agg[(long)j * Fdim + f], hi * cc);
  }
  atomicAdd(&agg[idx], acc);
}

__global__ void k_relu_to_f16(const float* __restrict__ in, _Float16* __restrict__ out, long n) {
  long i = (long)blockIdx.x * blockDim.x + threadIdx.x;
  if (i >= n) return;
  float v = in[i];
  out[i] = (_Float16)(v > 0.f ? v : 0.f);
}

// agg [N, D] node-major  ->  xg [B][D][L] (f32 and f16)
__global__ void k_layer3_finish(const float* __restrict__ agg, float* __restrict__ xg32,
                                _Float16* __restrict__ xg16) {
  long idx = (long)blockIdx.x * blockDim.x + threadIdx.x;
  long total = (long)BB * DD * LL;
  if (idx >= total) return;
  int b = (int)(idx / ((long)DD * LL));
  long r = idx - (long)b * DD * LL;
  int d = (int)(r / LL);
  int l = (int)(r - (long)d * LL);
  float v = agg[((long)b * LL + l) * DD + d];
  xg32[idx] = v;
  xg16[idx] = (_Float16)v;
}

// ---------------------------------------------------------------------------
// U-Net helpers
// ---------------------------------------------------------------------------
__global__ void k_maxpool2(const _Float16* __restrict__ x, _Float16* __restrict__ y,
                           long ntotal, int Lout) {
  long q = (long)blockIdx.x * blockDim.x + threadIdx.x;
  if (q >= ntotal) return;
  long bc = q / Lout;
  int l = (int)(q - bc * Lout);
  const _Float16* p = x + bc * (long)(2 * Lout) + 2 * l;
  _Float16 a = p[0], b = p[1];
  y[q] = (a > b) ? a : b;
}

__global__ void k_concat(const _Float16* __restrict__ a, const _Float16* __restrict__ u,
                         _Float16* __restrict__ out, int C1, int C2, int Ll) {
  long total = (long)BB * (C1 + C2) * Ll;
  long q = (long)blockIdx.x * blockDim.x + threadIdx.x;
  if (q >= total) return;
  int Ct = C1 + C2;
  int b = (int)(q / ((long)Ct * Ll));
  long r = q - (long)b * Ct * Ll;
  int c = (int)(r / Ll);
  int l = (int)(r - (long)c * Ll);
  out[q] = (c < C1) ? a[((long)b * C1 + c) * Ll + l]
                    : u[((long)b * C2 + (c - C1)) * Ll + l];
}

__global__ void k_mask(const _Float16* __restrict__ d1, const float* __restrict__ ow,
                       const float* __restrict__ ob, float* __restrict__ mask) {
  int idx = blockIdx.x * blockDim.x + threadIdx.x;
  if (idx >= BB * LL) return;
  int b = idx / LL, l = idx - b * LL;
  float acc = ob[0];
  const _Float16* p = d1 + ((long)b * 64) * LL + l;
  for (int c = 0; c < 64; ++c) acc += (float)p[(long)c * LL] * ow[c];
  mask[idx] = 1.f / (1.f + expf(-acc));
}

__global__ void k_maskmul(float* __restrict__ xg, const float* __restrict__ mask) {
  long total = (long)BB * DD * LL;
  long idx = (long)blockIdx.x * blockDim.x + threadIdx.x;
  if (idx >= total) return;
  int b = (int)(idx / ((long)DD * LL));
  int l = (int)(idx & (LL - 1));
  xg[idx] *= mask[b * LL + l];
}

// Overlap-add transposed-conv decoder (window 64, stride 32), truncated to T.
__global__ void k_decoder(const float* __restrict__ masked, const float* __restrict__ decw,
                          float* __restrict__ out) {
  long total = (long)BB * TT;
  long idx = (long)blockIdx.x * blockDim.x + threadIdx.x;
  if (idx >= total) return;
  int b = (int)(idx / TT);
  int t = (int)(idx - (long)b * TT);
  int lmax = t >> 5; if (lmax > LL - 1) lmax = LL - 1;
  int lmin = (t - 63) > 0 ? ((t - 63 + 31) >> 5) : 0;
  float acc = 0.f;
  const float* mb = masked + (long)b * DD * LL;
  for (int l = lmin; l <= lmax; ++l) {
    int k = t - (l << 5);            // in [0,63]
    const float* wk = decw + k;
    for (int d = 0; d < DD; ++d)
      acc += mb[(long)d * LL + l] * wk[(long)d * KWIN];
  }
  out[idx] = acc;
}

// ---------------------------------------------------------------------------
// Host orchestration
// ---------------------------------------------------------------------------
extern "C" void kernel_launch(void* const* d_in, const int* in_sizes, int n_in,
                              void* d_out, int out_size, void* d_ws, size_t ws_size,
                              hipStream_t stream) {
  (void)in_sizes; (void)n_in; (void)out_size; (void)ws_size;
  auto F32 = [&](int i) { return (const float*)d_in[i]; };

  // ---- input indices (setup_inputs dict order, flattened) ----
  const float* x_wave = F32(0);
  const float* enc_w  = F32(1);
  const float* W1 = F32(2); const float* b1 = F32(3);
  const float* W2 = F32(4); const float* b2 = F32(5);
  const float* W3 = F32(6); const float* b3 = F32(7);
  struct DC { const float *w1,*b1,*g1,*bb1,*m1,*v1,*w2,*b2,*g2,*bb2,*m2,*v2; };
  auto getDC = [&](int s) {
    DC d; d.w1=F32(s); d.b1=F32(s+1); d.g1=F32(s+2); d.bb1=F32(s+3); d.m1=F32(s+4); d.v1=F32(s+5);
    d.w2=F32(s+6); d.b2=F32(s+7); d.g2=F32(s+8); d.bb2=F32(s+9); d.m2=F32(s+10); d.v2=F32(s+11);
    return d;
  };
  DC inc = getDC(8), dn1 = getDC(20), dn2 = getDC(32), dn3 = getDC(44);
  const float* ct1w = F32(56); const float* ct1b = F32(57); DC u1dc = getDC(58);
  const float* ct2w = F32(70); const float* ct2b = F32(71); DC u2dc = getDC(72);
  const float* ct3w = F32(84); const float* ct3b = F32(85); DC u3dc = getDC(86);
  const float* outw = F32(98); const float* outb = F32(99);
  const float* decw = F32(100);
  float* out = (float*)d_out;

  // ---- workspace bump allocator ----
  char* base = (char*)d_ws;
  size_t off = 0;
  auto alloc = [&](size_t bytes) -> void* {
    off = (off + 255) & ~(size_t)255;
    void* p = base + off;
    off += bytes;
    return p;
  };
  auto cdiv = [](long a, long b) { return (a + b - 1) / b; };

  _Float16* xh    = (_Float16*)alloc((size_t)BB * TT * 2);
  _Float16* encwH = (_Float16*)alloc((size_t)DD * KWIN * 2);
  _Float16* W1h   = (_Float16*)alloc((size_t)DD * HH * 2);
  _Float16* W2h   = (_Float16*)alloc((size_t)HH * HH * 2);
  _Float16* W3h   = (_Float16*)alloc((size_t)HH * DD * 2);
  _Float16* F     = (_Float16*)alloc((size_t)NNODE * DD * 2);
  float*    sq    = (float*)   alloc((size_t)NNODE * 4);
  float*    Gbuf  = (float*)   alloc((size_t)LL * LL * 4);
  int*      nbr   = (int*)     alloc((size_t)NNODE * KNN8 * 4);
  float*    dinv  = (float*)   alloc((size_t)NNODE * 4);
  float*    hbuf  = (float*)   alloc((size_t)NNODE * DD * 4);
  float*    agg   = (float*)   alloc((size_t)NNODE * DD * 4);
  _Float16* xg1h  = (_Float16*)alloc((size_t)NNODE * HH * 2);
  _Float16* xg2h  = (_Float16*)alloc((size_t)NNODE * HH * 2);
  float*    xg32  = (float*)   alloc((size_t)BB * DD * LL * 4);
  _Float16* xg16  = (_Float16*)alloc((size_t)BB * DD * LL * 2);
  _Float16* x1  = (_Float16*)alloc((size_t)BB *  64 * 2048 * 2);
  _Float16* xp1 = (_Float16*)alloc((size_t)BB *  64 * 1024 * 2);
  _Float16* x2  = (_Float16*)alloc((size_t)BB * 128 * 1024 * 2);
  _Float16* xp2 = (_Float16*)alloc((size_t)BB * 128 *  512 * 2);
  _Float16* x3  = (_Float16*)alloc((size_t)BB * 256 *  512 * 2);
  _Float16* xp3 = (_Float16*)alloc((size_t)BB * 256 *  256 * 2);
  _Float16* x4  = (_Float16*)alloc((size_t)BB * 512 *  256 * 2);
  _Float16* u1  = (_Float16*)alloc((size_t)BB * 256 *  512 * 2);
  _Float16* c1  = (_Float16*)alloc((size_t)BB * 512 *  512 * 2);
  _Float16* d3  = (_Float16*)alloc((size_t)BB * 256 *  512 * 2);
  _Float16* u2  = (_Float16*)alloc((size_t)BB * 128 * 1024 * 2);
  _Float16* c2  = (_Float16*)alloc((size_t)BB * 256 * 1024 * 2);
  _Float16* d2  = (_Float16*)alloc((size_t)BB * 128 * 1024 * 2);
  _Float16* u3  = (_Float16*)alloc((size_t)BB *  64 * 2048 * 2);
  _Float16* c3  = (_Float16*)alloc((size_t)BB * 128 * 2048 * 2);
  _Float16* d1  = (_Float16*)alloc((size_t)BB *  64 * 2048 * 2);
  _Float16* tmp = (_Float16*)alloc((size_t)BB * 131072 * 2);   // per-DC intermediate
  float*    mask = (float*)  alloc((size_t)BB * LL * 4);

  // ---- weight prep ----
  struct ConvPack { _Float16* A; float* scale; float* shift; int cin; int cout; };
  auto prepConv = [&](const float* w, const float* bias, const float* g, const float* bb,
                      const float* m, const float* v, int cin, int cout) {
    ConvPack p; p.cin = cin; p.cout = cout;
    p.A = (_Float16*)alloc((size_t)cout * 3 * cin * 2);
    p.scale = (float*)alloc((size_t)cout * 4);
    p.shift = (float*)alloc((size_t)cout * 4);
    long n = (long)cout * 3 * cin;
    k_conv_w_reorder<<<cdiv(n, 256), 256, 0, stream>>>(w, p.A, cout, cin);
    k_bn_fold<<<cdiv(cout, 256), 256, 0, stream>>>(g, bb, m, v, bias, p.scale, p.shift, cout);
    return p;
  };
  auto prepDC = [&](const DC& d, int cin, int cout, ConvPack* pa, ConvPack* pb) {
    *pa = prepConv(d.w1, d.b1, d.g1, d.bb1, d.m1, d.v1, cin, cout);
    *pb = prepConv(d.w2, d.b2, d.g2, d.bb2, d.m2, d.v2, cout, cout);
  };
  ConvPack pInc1, pInc2, pD11, pD12, pD21, pD22, pD31, pD32, pU11, pU12, pU21, pU22, pU31, pU32;
  prepDC(inc, 512,  64, &pInc1, &pInc2);
  prepDC(dn1,  64, 128, &pD11, &pD12);
  prepDC(dn2, 128, 256, &pD21, &pD22);
  prepDC(dn3, 256, 512, &pD31, &pD32);
  prepDC(u1dc, 512, 256, &pU11, &pU12);
  prepDC(u2dc, 256, 128, &pU21, &pU22);
  prepDC(u3dc, 128,  64, &pU31, &pU32);

  auto prepCT = [&](const float* w, int cin, int cout, _Float16** A0, _Float16** A1) {
    *A0 = (_Float16*)alloc((size_t)cout * cin * 2);
    *A1 = (_Float16*)alloc((size_t)cout * cin * 2);
    long n = (long)cout * cin;
    k_ct_w_reorder<<<cdiv(n, 256), 256, 0, stream>>>(w, *A0, *A1, cin, cout);
  };
  _Float16 *ctA0_1, *ctA1_1, *ctA0_2, *ctA1_2, *ctA0_3, *ctA1_3;
  prepCT(ct1w, 512, 256, &ctA0_1, &ctA1_1);
  prepCT(ct2w, 256, 128, &ctA0_2, &ctA1_2);
  prepCT(ct3w, 128,  64, &ctA0_3, &ctA1_3);

  k_f32_to_f16<<<cdiv((long)BB * TT, 256), 256, 0, stream>>>(x_wave, xh, (long)BB * TT);
  k_f32_to_f16<<<cdiv(DD * KWIN, 256), 256, 0, stream>>>(enc_w, encwH, DD * KWIN);
  k_f32_to_f16<<<cdiv(DD * HH, 256), 256, 0, stream>>>(W1, W1h, DD * HH);
  k_f32_to_f16<<<cdiv(HH * HH, 256), 256, 0, stream>>>(W2, W2h, HH * HH);
  k_f32_to_f16<<<cdiv(HH * DD, 256), 256, 0, stream>>>(W3, W3h, HH * DD);

  // ---- 1) encoder: F[b,l,d] = windows(x) @ enc_w^T   (WMMA NT, NBLK=4) ----
  {
    dim3 g((unsigned)cdiv(DD / 64, 8), LL / 16, BB);
    k_gemm_nt<4><<<g, 256, 0, stream>>>(xh, STR, (long)TT, encwH, KWIN, 0L,
                                        nullptr, F, DD, (long)LL * DD, LL, DD, KWIN);
  }
  k_sq_rows<<<cdiv(NNODE, 256), 256, 0, stream>>>(F, sq, DD);

  // ---- 2) kNN graph: per-batch Gram matrix (WMMA NT, NBLK=4) + top-8 ----
  for (int b = 0; b < BB; ++b) {
    const _Float16* Fb = F + (long)b * LL * DD;
    dim3 g((unsigned)cdiv(LL / 64, 8), LL / 16, 1);
    k_gemm_nt<4><<<g, 256, 0, stream>>>(Fb, DD, 0L, Fb, DD, 0L,
                                        Gbuf, nullptr, LL, 0L, LL, LL, DD);
    k_topk<<<cdiv(LL, 256), 256, 0, stream>>>(Gbuf, sq, nbr, b);
  }

  // ---- 3) degrees / dinv ----
  k_fill_f32<<<cdiv(NNODE, 256), 256, 0, stream>>>(dinv, 9.f, NNODE);   // 8 (dst_k) + 1 (self)
  k_deg_accum<<<cdiv((long)NNODE * KNN8, 256), 256, 0, stream>>>(nbr, dinv, (long)NNODE * KNN8);
  k_rsqrt_inplace<<<cdiv(NNODE, 256), 256, 0, stream>>>(dinv, NNODE);

  // ---- 4) GCN layers (WMMA NN + atomic message passing) ----
  auto gcnLayer = [&](const _Float16* Xin, int Kdim, const _Float16* Wt, int Fdim,
                      const float* bias) {
    if (Fdim % 64 == 0) {
      dim3 g((unsigned)cdiv(Fdim / 64, 8), NNODE / 16, 1);
      k_gemm_nn<4><<<g, 256, 0, stream>>>(Xin, Kdim, 0L, Wt, Fdim, 0L,
                                          hbuf, nullptr, Fdim, 0L,
                                          nullptr, nullptr, 0, 1, 0, NNODE, Fdim, Kdim);
    } else {
      dim3 g((unsigned)cdiv(Fdim / 16, 8), NNODE / 16, 1);
      k_gemm_nn<1><<<g, 256, 0, stream>>>(Xin, Kdim, 0L, Wt, Fdim, 0L,
                                          hbuf, nullptr, Fdim, 0L,
                                          nullptr, nullptr, 0, 1, 0, NNODE, Fdim, Kdim);
    }
    long n = (long)NNODE * Fdim;
    k_gcn_init<<<cdiv(n, 256), 256, 0, stream>>>(hbuf, dinv, bias, agg, Fdim);
    k_gcn_edges<<<cdiv(n, 256), 256, 0, stream>>>(hbuf, dinv, nbr, agg, Fdim);
  };
  gcnLayer(F, DD, W1h, HH, b1);
  k_relu_to_f16<<<cdiv((long)NNODE * HH, 256), 256, 0, stream>>>(agg, xg1h, (long)NNODE * HH);
  gcnLayer(xg1h, HH, W2h, HH, b2);
  k_relu_to_f16<<<cdiv((long)NNODE * HH, 256), 256, 0, stream>>>(agg, xg2h, (long)NNODE * HH);
  gcnLayer(xg2h, HH, W3h, DD, b3);
  k_layer3_finish<<<cdiv((long)BB * DD * LL, 256), 256, 0, stream>>>(agg, xg32, xg16);

  // ---- 5) U-Net encoder path (implicit-GEMM WMMA convs, NBLK=4) ----
  auto conv3 = [&](const _Float16* X, const ConvPack& p, _Float16* Y, int Llen) {
    dim3 g((unsigned)cdiv(Llen / 64, 8), p.cout / 16, BB);
    k_conv3<4><<<g, 256, 0, stream>>>(X, (long)p.cin * Llen, p.A, p.scale, p.shift,
                                      Y, (long)p.cout * Llen, p.cin, p.cout, Llen);
  };
  auto runDC = [&](const _Float16* X, const ConvPack& pa, const ConvPack& pb,
                   _Float16* Y, int Llen) {
    conv3(X, pa, tmp, Llen);
    conv3(tmp, pb, Y, Llen);
  };
  runDC(xg16, pInc1, pInc2, x1, 2048);
  k_maxpool2<<<cdiv((long)BB * 64 * 1024, 256), 256, 0, stream>>>(x1, xp1, (long)BB * 64 * 1024, 1024);
  runDC(xp1, pD11, pD12, x2, 1024);
  k_maxpool2<<<cdiv((long)BB * 128 * 512, 256), 256, 0, stream>>>(x2, xp2, (long)BB * 128 * 512, 512);
  runDC(xp2, pD21, pD22, x3, 512);
  k_maxpool2<<<cdiv((long)BB * 256 * 256, 256), 256, 0, stream>>>(x3, xp3, (long)BB * 256 * 256, 256);
  runDC(xp3, pD31, pD32, x4, 256);

  // ---- 6) U-Net decoder path (convT = 2 strided-column WMMA GEMMs, NBLK=4) ----
  auto convT = [&](const _Float16* A0, const _Float16* A1, const float* bias,
                   const _Float16* X, _Float16* U, int cin, int cout, int Ls) {
    dim3 g((unsigned)cdiv(Ls / 64, 8), cout / 16, BB);
    k_gemm_nn<4><<<g, 256, 0, stream>>>(A0, cin, 0L, X, Ls, (long)cin * Ls,
                                        nullptr, U, 2 * Ls, (long)cout * 2 * Ls,
                                        nullptr, bias, 0, 2, 0, cout, Ls, cin);
    k_gemm_nn<4><<<g, 256, 0, stream>>>(A1, cin, 0L, X, Ls, (long)cin * Ls,
                                        nullptr, U, 2 * Ls, (long)cout * 2 * Ls,
                                        nullptr, bias, 0, 2, 1, cout, Ls, cin);
  };
  convT(ctA0_1, ctA1_1, ct1b, x4, u1, 512, 256, 256);
  k_concat<<<cdiv((long)BB * 512 * 512, 256), 256, 0, stream>>>(x3, u1, c1, 256, 256, 512);
  runDC(c1, pU11, pU12, d3, 512);
  convT(ctA0_2, ctA1_2, ct2b, d3, u2, 256, 128, 512);
  k_concat<<<cdiv((long)BB * 256 * 1024, 256), 256, 0, stream>>>(x2, u2, c2, 128, 128, 1024);
  runDC(c2, pU21, pU22, d2, 1024);
  convT(ctA0_3, ctA1_3, ct3b, d2, u3, 128, 64, 1024);
  k_concat<<<cdiv((long)BB * 128 * 2048, 256), 256, 0, stream>>>(x1, u3, c3, 64, 64, 2048);
  runDC(c3, pU31, pU32, d1, 2048);

  // ---- 7) mask, apply, overlap-add decode ----
  k_mask<<<cdiv(BB * LL, 256), 256, 0, stream>>>(d1, outw, outb, mask);
  k_maskmul<<<cdiv((long)BB * DD * LL, 256), 256, 0, stream>>>(xg32, mask);
  k_decoder<<<cdiv((long)BB * TT, 256), 256, 0, stream>>>(xg32, decw, out);
}
